// GroupedQueryAttention_62775241998539
// MI455X (gfx1250) — matrix operand order repaired and checked by reading, not compiled, not run on hardware
//
#include <hip/hip_runtime.h>
#include <hip/hip_bf16.h>
#include <math.h>

#define BATCH 2
#define SEQ   2048
#define HID   4096
#define NH    32
#define NKV   8
#define HD    128
#define GQ    (NH / NKV)          // 4
#define MROWS (BATCH * SEQ)       // 4096

typedef __attribute__((ext_vector_type(16))) __bf16 v16bf;
typedef __attribute__((ext_vector_type(8)))  float  v8f;

union ABFrag {
  v16bf v;
  uint4 u4[2];
};

__device__ __forceinline__ unsigned short f2bf(float f) {
  unsigned int u = __float_as_uint(f);
  u += 0x7FFFu + ((u >> 16) & 1u);           // round-to-nearest-even
  return (unsigned short)(u >> 16);
}
__device__ __forceinline__ float bf2f(unsigned short h) {
  return __uint_as_float(((unsigned int)h) << 16);
}

__device__ __forceinline__ v8f wmma_bf16(const ABFrag& a, const ABFrag& b, v8f c) {
  // D = A(16x32 bf16) * B(32x16 bf16) + C(16x16 f32)
  return __builtin_amdgcn_wmma_f32_16x16x32_bf16(false, a.v, false, b.v,
                                                 (short)0, c, false, false);
}

// ---------------------------------------------------------------------------
// fp32 -> bf16 elementwise
// ---------------------------------------------------------------------------
__global__ void f32_to_bf16(const float* __restrict__ in,
                            unsigned short* __restrict__ out, size_t n) {
  size_t i = (size_t)blockIdx.x * blockDim.x + threadIdx.x;
  if (i < n) out[i] = f2bf(in[i]);
}

// ---------------------------------------------------------------------------
// C(MxN) = A(MxK) * W(NxK)^T     (all bf16 in, f32 accumulate)
// Block: 256 threads = 8 waves arranged 2x4; block tile 128x256.
// Wave tile 64x64 = 4x4 WMMA tiles -> 16 WMMAs per K-step (K-step 32).
// Double-buffered LDS tiles, one barrier per K-step, global loads of the
// next tile issued before the current step's WMMAs.
// ---------------------------------------------------------------------------
template <bool OUT_BF16>
__global__ __launch_bounds__(256) void gemm_bf16_wmma(
    const unsigned short* __restrict__ A,   // M x K, row-major
    const unsigned short* __restrict__ W,   // N x K, row-major
    void* __restrict__ C, int M, int N, int K)
{
  __shared__ __attribute__((aligned(16))) unsigned short As[2][128 * 40];
  __shared__ __attribute__((aligned(16))) unsigned short Bs[2][256 * 40];

  const int tid  = threadIdx.x;
  const int lane = tid & 31;
  const int wave = tid >> 5;
  const int wm   = wave >> 2;        // 0..1  -> 64-row sub-tile
  const int wn   = wave & 3;         // 0..3  -> 64-col sub-tile
  const int hf   = lane >> 4;        // half of wave
  const int l16  = lane & 15;
  const int mBase = blockIdx.x * 128;
  const int nBase = blockIdx.y * 256;

  // cooperative global->LDS load mapping
  const int arow = tid >> 1, acol = (tid & 1) * 16;   // A: 128 rows x 32 cols
  const unsigned short* Ag = A + (size_t)(mBase + arow) * K + acol;
  const unsigned short* Wg = W + (size_t)(nBase + tid) * K;  // B: 256 rows x 32
  const int aoff = arow * 40 + acol;
  const int boff = tid * 40;

  v8f acc[4][4];
  for (int mt = 0; mt < 4; ++mt)
    for (int nt = 0; nt < 4; ++nt) acc[mt][nt] = {};

  // prologue: tile 0 -> LDS buffer 0
  uint4 ra0 = *(const uint4*)(Ag);
  uint4 ra1 = *(const uint4*)(Ag + 8);
  uint4 rb[4];
  for (int c = 0; c < 4; ++c) rb[c] = *(const uint4*)(Wg + c * 8);
  *(uint4*)&As[0][aoff]     = ra0;
  *(uint4*)&As[0][aoff + 8] = ra1;
  for (int c = 0; c < 4; ++c) *(uint4*)&Bs[0][boff + c * 8] = rb[c];
  __syncthreads();

  int cur = 0;
  for (int kk = 0; kk < K; kk += 32) {
    const bool more = (kk + 32) < K;
    if (more) {   // start next tile's global loads early (overlap with WMMA)
      ra0 = *(const uint4*)(Ag + kk + 32);
      ra1 = *(const uint4*)(Ag + kk + 40);
      for (int c = 0; c < 4; ++c) rb[c] = *(const uint4*)(Wg + kk + 32 + c * 8);
    }

    ABFrag af[4], bfr[4];
    for (int mt = 0; mt < 4; ++mt) {
      // A frag: lanes 0-15 -> row=l16, K {0..7,16..23}; lanes 16-31 K +8
      const unsigned short* p = &As[cur][(wm * 64 + mt * 16 + l16) * 40 + hf * 8];
      af[mt].u4[0] = *(const uint4*)p;
      af[mt].u4[1] = *(const uint4*)(p + 16);
    }
    for (int nt = 0; nt < 4; ++nt) {
      // B frag: lane -> column, K {0..7,16..23}; lanes 16-31 K +8
      const unsigned short* p = &Bs[cur][(wn * 64 + nt * 16 + l16) * 40 + hf * 8];
      bfr[nt].u4[0] = *(const uint4*)p;
      bfr[nt].u4[1] = *(const uint4*)(p + 16);
    }
    for (int mt = 0; mt < 4; ++mt)
      for (int nt = 0; nt < 4; ++nt)
        acc[mt][nt] = wmma_bf16(af[mt], bfr[nt], acc[mt][nt]);

    if (more) {
      const int nxt = cur ^ 1;
      *(uint4*)&As[nxt][aoff]     = ra0;
      *(uint4*)&As[nxt][aoff + 8] = ra1;
      for (int c = 0; c < 4; ++c) *(uint4*)&Bs[nxt][boff + c * 8] = rb[c];
      __syncthreads();
      cur = nxt;
    }
  }

  // C/D layout: lanes 0-15: (M=j, N=lane); lanes 16-31: (M=j+8, N=lane-16)
  for (int mt = 0; mt < 4; ++mt)
    for (int nt = 0; nt < 4; ++nt)
      for (int j = 0; j < 8; ++j) {
        int r  = mBase + wm * 64 + mt * 16 + hf * 8 + j;
        int cI = nBase + wn * 64 + nt * 16 + l16;
        if (OUT_BF16)
          ((unsigned short*)C)[(size_t)r * N + cI] = f2bf(acc[mt][nt][j]);
        else
          ((float*)C)[(size_t)r * N + cI] = acc[mt][nt][j];
      }
}

// ---------------------------------------------------------------------------
// RoPE in-place on (B*S, heads, 128) bf16.  One thread per rotation pair.
// scale folds the attention 1/sqrt(D) into Q.
// ---------------------------------------------------------------------------
__global__ void rope_apply(unsigned short* __restrict__ X, int heads, float scale) {
  size_t idx = (size_t)blockIdx.x * blockDim.x + threadIdx.x;
  size_t total = (size_t)MROWS * heads * 64;
  if (idx >= total) return;
  int j = (int)(idx & 63);
  size_t t = idx >> 6;
  int head = (int)(t % heads);
  size_t bs = t / heads;              // b*SEQ + s
  int s = (int)(bs % SEQ);
  float inv = __powf(10000.0f, -(float)j * (1.0f / 64.0f));
  float ang = (float)s * inv;
  float sn, cs;
  __sincosf(ang, &sn, &cs);
  unsigned short* p = X + (bs * (size_t)heads + head) * (size_t)HD;
  float x1 = bf2f(p[j]);
  float x2 = bf2f(p[j + 64]);
  p[j]      = f2bf((x1 * cs - x2 * sn) * scale);
  p[j + 64] = f2bf((x2 * cs + x1 * sn) * scale);
}

// ---------------------------------------------------------------------------
// V (B*S, NKV*HD)  ->  Vt (B, NKV, HD, S)   (keys contiguous for B-fragments)
// ---------------------------------------------------------------------------
__global__ void transpose_v(const unsigned short* __restrict__ V,
                            unsigned short* __restrict__ Vt) {
  size_t idx = (size_t)blockIdx.x * blockDim.x + threadIdx.x;
  size_t total = (size_t)MROWS * NKV * HD;
  if (idx >= total) return;
  int d = (int)(idx % HD);
  size_t t = idx / HD;
  int kh = (int)(t % NKV);
  size_t bs = t / NKV;
  int s = (int)(bs % SEQ);
  int b = (int)(bs / SEQ);
  Vt[((size_t)(b * NKV + kh) * HD + d) * SEQ + s] = V[idx];
}

// ---------------------------------------------------------------------------
// Flash attention (causal, GQA).  grid = (S/64, NH, B), 128 threads = 4 waves.
// Each wave owns 16 query rows; streams 32-key chunks with online softmax.
// All 16 fragment loads of a chunk are issued before any WMMA so the
// compiler can clause them and overlap V loads with score WMMAs + softmax.
// ---------------------------------------------------------------------------
__global__ __launch_bounds__(128) void flash_attn_gqa(
    const unsigned short* __restrict__ Q,    // (B*S, NH*HD), RoPE'd & prescaled
    const unsigned short* __restrict__ Kb,   // (B*S, NKV*HD), RoPE'd
    const unsigned short* __restrict__ Vt,   // (B, NKV, HD, S)
    unsigned short* __restrict__ O)          // (B*S, NH*HD)
{
  const int lane = threadIdx.x & 31;
  const int wave = threadIdx.x >> 5;
  const int hf   = lane >> 4;
  const int l16  = lane & 15;
  const int h  = blockIdx.y;
  const int b  = blockIdx.z;
  const int kh = h / GQ;
  const int qBase = blockIdx.x * 64 + wave * 16;

  // per-wave P staging (16x32, padded rows for bank-conflict-free ds_load)
  __shared__ __attribute__((aligned(16))) unsigned short pstage[4][16 * 40];

  // Q fragments: 4 chunks of K=32 over D=128 (kept resident)
  ABFrag qf[4];
  const unsigned short* qrow = Q + ((size_t)(b * SEQ + qBase + l16) * NH + h) * HD;
  for (int dc = 0; dc < 4; ++dc) {
    const uint4* p = (const uint4*)(qrow + dc * 32 + hf * 16);
    qf[dc].u4[0] = p[0];
    qf[dc].u4[1] = p[1];
  }

  v8f o[8];                                  // 16 x 128 f32 accumulator
  for (int t = 0; t < 8; ++t) o[t] = {};
  float m[8], lsum[8];
  for (int j = 0; j < 8; ++j) { m[j] = -3.0e38f; lsum[j] = 0.0f; }

  const int kmax = qBase + 15;               // causal horizon for this wave
  const size_t krow = (size_t)NKV * HD;
  const unsigned short* vbase =
      Vt + ((size_t)(b * NKV + kh) * HD + l16) * SEQ + hf * 16;

  for (int c0 = 0; c0 <= kmax; c0 += 32) {
    // ---- issue ALL chunk loads up front: 16 K-frag + 16 V-frag b128 ----
    const unsigned short* kp0 =
        Kb + (size_t)(b * SEQ + c0 + l16) * krow + kh * HD + hf * 16;
    const unsigned short* kp1 = kp0 + 16 * krow;
    __builtin_prefetch(kp0 + 32 * krow, 0, 0);          // next chunk's K
    ABFrag bk[8];
    for (int dc = 0; dc < 4; ++dc) {
      bk[dc].u4[0]     = ((const uint4*)(kp0 + dc * 32))[0];
      bk[dc].u4[1]     = ((const uint4*)(kp0 + dc * 32))[1];
      bk[4 + dc].u4[0] = ((const uint4*)(kp1 + dc * 32))[0];
      bk[4 + dc].u4[1] = ((const uint4*)(kp1 + dc * 32))[1];
    }
    ABFrag bv[8];
    const unsigned short* vb = vbase + c0;
    for (int t = 0; t < 8; ++t) {
      const unsigned short* vp = vb + (size_t)t * 16 * SEQ;
      bv[t].u4[0] = ((const uint4*)vp)[0];
      bv[t].u4[1] = ((const uint4*)vp)[1];
    }

    // ---- scores S = Q K^T for 32 keys (two independent WMMA chains) ----
    v8f s0 = {}, s1 = {};
    for (int dc = 0; dc < 4; ++dc) {
      s0 = wmma_bf16(qf[dc], bk[dc], s0);
      s1 = wmma_bf16(qf[dc], bk[4 + dc], s1);
    }

    // ---- causal mask (only chunks that touch the diagonal) ----
    if (c0 + 31 > qBase) {
      for (int j = 0; j < 8; ++j) {
        int q = qBase + hf * 8 + j;
        if (c0 + l16 > q)      s0[j] = -1.0e30f;
        if (c0 + 16 + l16 > q) s1[j] = -1.0e30f;
      }
    }

    // ---- online softmax: rows live in 16-lane halves ----
    float al[8];
    for (int j = 0; j < 8; ++j) {
      float mx = fmaxf(s0[j], s1[j]);
      for (int d = 1; d < 16; d <<= 1) mx = fmaxf(mx, __shfl_xor(mx, d, 32));
      float mn = fmaxf(m[j], mx);
      float a  = __expf(m[j] - mn);
      m[j] = mn;
      float p0 = __expf(s0[j] - mn);
      float p1 = __expf(s1[j] - mn);
      float rs = p0 + p1;
      for (int d = 1; d < 16; d <<= 1) rs += __shfl_xor(rs, d, 32);
      lsum[j] = lsum[j] * a + rs;
      al[j] = a;
      pstage[wave][(hf * 8 + j) * 40 + l16]      = f2bf(p0);   // key c0+l16
      pstage[wave][(hf * 8 + j) * 40 + l16 + 16] = f2bf(p1);   // key c0+16+l16
    }
    for (int t = 0; t < 8; ++t)
      for (int j = 0; j < 8; ++j) o[t][j] *= al[j];

    // LDS handoff: DS ops are in-order per wave; make it explicit
    asm volatile("s_wait_dscnt 0x0" ::: "memory");

    // ---- P (16x32) as A-fragment from LDS ----
    ABFrag pf;
    const unsigned short* pr = &pstage[wave][l16 * 40 + hf * 16];
    pf.u4[0] = *(const uint4*)pr;
    pf.u4[1] = *(const uint4*)(pr + 8);

    // ---- O += P V : 8 independent WMMA accumulations ----
    for (int t = 0; t < 8; ++t)
      o[t] = wmma_bf16(pf, bv[t], o[t]);
  }

  // ---- normalize and store (B,S,H*D) bf16 ----
  for (int t = 0; t < 8; ++t)
    for (int j = 0; j < 8; ++j) {
      int q = qBase + hf * 8 + j;
      float val = o[t][j] / lsum[j];
      O[(size_t)(b * SEQ + q) * (NH * HD) + h * HD + t * 16 + l16] = f2bf(val);
    }
}

// ---------------------------------------------------------------------------
// host-side launch
// ---------------------------------------------------------------------------
extern "C" void kernel_launch(void* const* d_in, const int* in_sizes, int n_in,
                              void* d_out, int out_size, void* d_ws, size_t ws_size,
                              hipStream_t stream) {
  (void)in_sizes; (void)n_in; (void)out_size; (void)ws_size;
  const float* hs = (const float*)d_in[0];
  // d_in[1] = attention_mask: pure causal; synthesized in-kernel, not read.
  const float* wq = (const float*)d_in[2];
  const float* wk = (const float*)d_in[3];
  const float* wv = (const float*)d_in[4];
  const float* wo = (const float*)d_in[5];
  float* out = (float*)d_out;

  char* ws = (char*)d_ws;
  size_t off = 0;
  auto alloc = [&](size_t elems) -> unsigned short* {
    unsigned short* p = (unsigned short*)(ws + off);
    off += (elems * sizeof(unsigned short) + 255) & ~(size_t)255;
    return p;
  };
  const size_t nX  = (size_t)MROWS * HID;          // 16.8M
  const size_t nWq = (size_t)HID * HID;            // 16.8M
  const size_t nWk = (size_t)(NKV * HD) * HID;     // 4.2M
  unsigned short* Xb  = alloc(nX);
  unsigned short* Wqb = alloc(nWq);
  unsigned short* Wkb = alloc(nWk);
  unsigned short* Wvb = alloc(nWk);
  unsigned short* Wob = alloc(nWq);
  unsigned short* Qb  = alloc((size_t)MROWS * NH * HD);
  unsigned short* Kbf = alloc((size_t)MROWS * NKV * HD);
  unsigned short* Vbf = alloc((size_t)MROWS * NKV * HD);
  unsigned short* Vtb = alloc((size_t)MROWS * NKV * HD);
  unsigned short* Ob  = alloc((size_t)MROWS * NH * HD);

  auto cvt = [&](const float* src, unsigned short* dst, size_t n) {
    f32_to_bf16<<<dim3((unsigned)((n + 255) / 256)), dim3(256), 0, stream>>>(src, dst, n);
  };
  cvt(hs, Xb, nX);
  cvt(wq, Wqb, nWq);
  cvt(wk, Wkb, nWk);
  cvt(wv, Wvb, nWk);
  cvt(wo, Wob, nWq);

  // projections: Q (4096x4096), K/V (4096x1024); block tile 128x256
  gemm_bf16_wmma<true><<<dim3(MROWS / 128, HID / 256), 256, 0, stream>>>(
      Xb, Wqb, Qb, MROWS, HID, HID);
  gemm_bf16_wmma<true><<<dim3(MROWS / 128, (NKV * HD) / 256), 256, 0, stream>>>(
      Xb, Wkb, Kbf, MROWS, NKV * HD, HID);
  gemm_bf16_wmma<true><<<dim3(MROWS / 128, (NKV * HD) / 256), 256, 0, stream>>>(
      Xb, Wvb, Vbf, MROWS, NKV * HD, HID);

  // RoPE (fold softmax scale into Q)
  float qscale = 1.0f / sqrtf((float)HD);
  rope_apply<<<dim3((unsigned)(((size_t)MROWS * NH * 64 + 255) / 256)), 256, 0, stream>>>(
      Qb, NH, qscale);
  rope_apply<<<dim3((unsigned)(((size_t)MROWS * NKV * 64 + 255) / 256)), 256, 0, stream>>>(
      Kbf, NKV, 1.0f);

  transpose_v<<<dim3((unsigned)(((size_t)MROWS * NKV * HD + 255) / 256)), 256, 0, stream>>>(
      Vbf, Vtb);

  flash_attn_gqa<<<dim3(SEQ / 64, NH, BATCH), 128, 0, stream>>>(Qb, Kbf, Vtb, Ob);

  // output projection, f32 out
  gemm_bf16_wmma<false><<<dim3(MROWS / 128, HID / 256), 256, 0, stream>>>(
      Ob, Wob, out, MROWS, HID, HID);
}
